// Seq2SeqDecoder_17798344475047
// MI455X (gfx1250) — compile-verified, matrix operand order checked by reading, not run on hardware
//
#include <hip/hip_runtime.h>
#include <hip/hip_bf16.h>
#include <cstdint>
#include <cstddef>

// ---------------------------------------------------------------------------
// CDNA5 (gfx1250) GRU seq2seq decoder.
// All GEMMs via v_wmma_f32_16x16x32_bf16 (bf16 operands, f32 accumulate).
// ---------------------------------------------------------------------------

typedef __bf16 bf16_t;
typedef __attribute__((ext_vector_type(16))) __bf16 v16bf;
typedef __attribute__((ext_vector_type(8)))  __bf16 v8bf;
typedef __attribute__((ext_vector_type(8)))  float  v8f;

#define B_   32
#define T_   64
#define V_   32000
#define E_   512
#define H_   1024
#define G_   3072      // 3*H
#define EH_  1536      // E + H
#define BT_  2048      // B*T
#define BH_  32768     // B*H

// ---------------------------------------------------------------------------
// Embedding gather + context concat -> bf16 xin [B*T, EH]
// row index = b*T + t ; cols [0,512) from emb[X], cols [512,1536) from state[1]
// ---------------------------------------------------------------------------
__global__ void embed_concat_kernel(const int* __restrict__ X,
                                    const float* __restrict__ state,
                                    const float* __restrict__ emb,
                                    bf16_t* __restrict__ xin)
{
    size_t idx = (size_t)blockIdx.x * blockDim.x + threadIdx.x;   // BT_*EH_
    int    c   = (int)(idx % EH_);
    size_t bt  = idx / EH_;
    int    b   = (int)(bt / T_);
    float v;
    if (c < E_) {
        int tok = X[bt];
        v = emb[(size_t)tok * E_ + c];
    } else {
        // ctx = state[-1] = state[1]
        v = state[(size_t)(B_ + b) * H_ + (c - E_)];
    }
    xin[idx] = (bf16_t)v;
}

// ---------------------------------------------------------------------------
// Initial hidden states: h0 = state[0], h1 = state[1]  (f32 + bf16 copies)
// ---------------------------------------------------------------------------
__global__ void init_states_kernel(const float* __restrict__ state,
                                   float* __restrict__ h0, float* __restrict__ h1,
                                   bf16_t* __restrict__ h0b, bf16_t* __restrict__ h1b)
{
    int idx = blockIdx.x * blockDim.x + threadIdx.x;   // BH_
    float s0 = state[idx];
    float s1 = state[BH_ + idx];
    h0[idx] = s0;  h0b[idx] = (bf16_t)s0;
    h1[idx] = s1;  h1b[idx] = (bf16_t)s1;
}

// ---------------------------------------------------------------------------
// Pack a row-major f32 weight W[K,N] into bf16 WMMA B-fragment tiles.
// Tile = 32 (K) x 16 (N). Packed layout: [kt][nt][lane 0..31][elem 0..15].
// Lane n (and n+16) hold column n with the ISA 16-bit K striping:
//   lane<16 : e<8 -> K=e       ; e>=8 -> K=e+8    (K 0..7, 16..23)
//   lane>=16: e<8 -> K=e+8     ; e>=8 -> K=e+16   (K 8..15, 24..31)
// One lane's fragment = contiguous 32 bytes -> coalesced 1KB tile loads.
// ---------------------------------------------------------------------------
__global__ void pack_b_kernel(const float* __restrict__ W,
                              bf16_t* __restrict__ out,
                              int K, int N)
{
    size_t idx = (size_t)blockIdx.x * blockDim.x + threadIdx.x;
    size_t total = (size_t)K * N;
    if (idx >= total) return;
    int    e    = (int)(idx & 15);
    int    lane = (int)((idx >> 4) & 31);
    size_t tile = idx >> 9;
    int    NT   = N >> 4;
    int    nt   = (int)(tile % NT);
    int    kt   = (int)(tile / NT);
    int    k    = (e & 7) + ((e >> 3) << 4) + ((lane >> 4) << 3);
    int    col  = lane & 15;
    out[idx] = (bf16_t)W[(size_t)(kt * 32 + k) * N + (nt << 4) + col];
}

// ---------------------------------------------------------------------------
// bf16 WMMA GEMM:  C[M,N] = A[M,K] @ Bpacked + bias   (f32 accumulate)
// One wave computes a 16x128 output strip (1 M-tile x 8 N-tiles), reusing the
// A fragment across 8 WMMAs per K-step. Grid is sized exactly -> EXEC stays
// all-ones (WMMA requirement). Unconditional speculative global_prefetch_b8
// of the next K-step's B tiles keeps the loop branch-free.
// ---------------------------------------------------------------------------
__global__ void gemm_bf16_wmma_kernel(const bf16_t* __restrict__ A, int lda,
                                      const bf16_t* __restrict__ Bp,
                                      const float* __restrict__ bias,
                                      float* __restrict__ C, int ldc,
                                      int M, int N, int K)
{
    const int lane    = threadIdx.x & 31;
    const int gw      = (int)((blockIdx.x * blockDim.x + threadIdx.x) >> 5);
    const int ntiles  = N >> 4;
    const int ngroups = ntiles >> 3;           // 8 N-tiles per wave
    const int mt      = gw / ngroups;
    const int ng      = gw % ngroups;
    if (mt >= (M >> 4)) return;                // never taken (exact grid)
    const int m0   = mt << 4;
    const int n0   = ng << 7;                  // 8 * 16
    const int nt0  = n0 >> 4;

    const int row  = m0 + (lane & 15);
    const int koff = (lane >> 4) << 3;         // 0 or 8

    v8f acc[8];
#pragma unroll
    for (int j = 0; j < 8; ++j) acc[j] = (v8f){};

    const size_t bstride_k = (size_t)ntiles * 512;   // elements per K-step
    const bf16_t* bp = Bp + ((size_t)nt0 * 32 + lane) * 16;
    const bf16_t* ap = A + (size_t)row * lda + koff;

    const int ktiles = K >> 5;
    for (int kt = 0; kt < ktiles; ++kt) {
        // ---- A fragment (row-major bf16, ISA K-striping built on the fly) ----
        v8bf alo = *(const v8bf*)(ap);
        v8bf ahi = *(const v8bf*)(ap + 16);
        v16bf a;
#pragma unroll
        for (int e = 0; e < 8; ++e) { a[e] = alo[e]; a[8 + e] = ahi[e]; }

        // ---- B fragments (pre-packed, contiguous 32B per lane per tile) ----
        v16bf bfrag[8];
#pragma unroll
        for (int j = 0; j < 8; ++j)
            bfrag[j] = *(const v16bf*)(bp + (size_t)j * 512);

        // speculative prefetch of next K-step's B tiles (dropped if past end)
        __builtin_prefetch((const void*)(bp + bstride_k), 0, 1);

#pragma unroll
        for (int j = 0; j < 8; ++j)
            acc[j] = __builtin_amdgcn_wmma_f32_16x16x32_bf16(
                false, a, false, bfrag[j], (short)0, acc[j], false, false);

        ap += 32;
        bp += bstride_k;
    }

    // ---- store C (+bias): lane<16 -> rows m0+0..7 ; lane>=16 -> m0+8..15 ----
    const int col   = lane & 15;
    const int rbase = (lane < 16) ? 0 : 8;
    float bi[8];
#pragma unroll
    for (int j = 0; j < 8; ++j) bi[j] = bias[n0 + 16 * j + col];
    float* crow = C + (size_t)(m0 + rbase) * ldc + n0 + col;
#pragma unroll
    for (int i = 0; i < 8; ++i) {
        float* cr = crow + (size_t)i * ldc;
#pragma unroll
        for (int j = 0; j < 8; ++j)
            cr[16 * j] = acc[j][i] + bi[j];
    }
}

// ---------------------------------------------------------------------------
// Fused GRU gates (Keras reset_after=True):
//   z = sig(xz+iz); r = sig(xr+ir); hh = tanh(xh + r*ih)
//   h_new = z*h + (1-z)*hh
// mx rows may be strided (layer0 reads a T-slice of the big precompute).
// ---------------------------------------------------------------------------
__global__ void gru_gates_kernel(const float* __restrict__ mx, int mx_row_stride,
                                 const float* __restrict__ mi,
                                 const float* __restrict__ h_in,
                                 float* __restrict__ h_out,
                                 bf16_t* __restrict__ h_out_bf,
                                 bf16_t* __restrict__ seq_out, int t)
{
    int idx = blockIdx.x * blockDim.x + threadIdx.x;   // BH_
    int b = idx >> 10;
    int j = idx & (H_ - 1);
    const float* mxb = mx + (size_t)b * mx_row_stride;
    const float* mib = mi + (size_t)b * G_;
    float xz = mxb[j], xr = mxb[H_ + j], xh = mxb[2 * H_ + j];
    float iz = mib[j], ir = mib[H_ + j], ih = mib[2 * H_ + j];
    float z  = 1.0f / (1.0f + __expf(-(xz + iz)));
    float r  = 1.0f / (1.0f + __expf(-(xr + ir)));
    float hh = tanhf(xh + r * ih);
    float h  = z * h_in[idx] + (1.0f - z) * hh;
    h_out[idx]    = h;
    h_out_bf[idx] = (bf16_t)h;
    if (seq_out) {
        seq_out[((size_t)b * T_ + t) * H_ + j] = (bf16_t)h;
    }
}

// ---------------------------------------------------------------------------
// Final states -> d_out tail: [h0f ; h1f], each B*H f32.
// ---------------------------------------------------------------------------
__global__ void finalize_states_kernel(const float* __restrict__ h0,
                                       const float* __restrict__ h1,
                                       float* __restrict__ out)
{
    int idx = blockIdx.x * blockDim.x + threadIdx.x;   // 2*BH_
    out[idx] = (idx < BH_) ? h0[idx] : h1[idx - BH_];
}

// ---------------------------------------------------------------------------
// Host side
// ---------------------------------------------------------------------------
static inline size_t align256(size_t x) { return (x + 255) & ~(size_t)255; }

extern "C" void kernel_launch(void* const* d_in, const int* in_sizes, int n_in,
                              void* d_out, int out_size, void* d_ws, size_t ws_size,
                              hipStream_t stream) {
    (void)in_sizes; (void)n_in; (void)out_size; (void)ws_size;

    const int*   X     = (const int*)  d_in[0];
    const float* state = (const float*)d_in[1];
    const float* emb   = (const float*)d_in[2];
    const float* k0    = (const float*)d_in[3];
    const float* rk0   = (const float*)d_in[4];
    const float* b0    = (const float*)d_in[5];   // [2, 3H]
    const float* k1    = (const float*)d_in[6];
    const float* rk1   = (const float*)d_in[7];
    const float* b1    = (const float*)d_in[8];   // [2, 3H]
    const float* Wd    = (const float*)d_in[9];
    const float* bd    = (const float*)d_in[10];

    float* out = (float*)d_out;                   // logits [BT,V] then states [2,B,H]

    // ---- carve workspace ----
    char* ws = (char*)d_ws;  size_t off = 0;
    auto carve = [&](size_t bytes) { void* p = ws + off; off += align256(bytes); return p; };

    bf16_t* xin  = (bf16_t*)carve((size_t)BT_ * EH_ * 2);
    bf16_t* k0p  = (bf16_t*)carve((size_t)EH_ * G_ * 2);
    bf16_t* rk0p = (bf16_t*)carve((size_t)H_  * G_ * 2);
    bf16_t* k1p  = (bf16_t*)carve((size_t)H_  * G_ * 2);
    bf16_t* rk1p = (bf16_t*)carve((size_t)H_  * G_ * 2);
    bf16_t* Wdp  = (bf16_t*)carve((size_t)H_  * V_ * 2);
    float*  mx0  = (float*) carve((size_t)BT_ * G_ * 4);
    float*  mi0  = (float*) carve((size_t)B_  * G_ * 4);
    float*  mx1  = (float*) carve((size_t)B_  * G_ * 4);
    float*  mi1  = (float*) carve((size_t)B_  * G_ * 4);
    float*  h0   = (float*) carve((size_t)BH_ * 4);
    float*  h1   = (float*) carve((size_t)BH_ * 4);
    bf16_t* h0b  = (bf16_t*)carve((size_t)BH_ * 2);
    bf16_t* h1b  = (bf16_t*)carve((size_t)BH_ * 2);
    bf16_t* seq  = (bf16_t*)carve((size_t)BT_ * H_ * 2);

    const int TPB = 256;

    // ---- 1) embedding + concat, initial states ----
    {
        size_t n = (size_t)BT_ * EH_;
        embed_concat_kernel<<<(unsigned)((n + TPB - 1) / TPB), TPB, 0, stream>>>(
            X, state, emb, xin);
        init_states_kernel<<<BH_ / TPB, TPB, 0, stream>>>(state, h0, h1, h0b, h1b);
    }

    // ---- 2) pack weights to WMMA tile layout (bf16) ----
    {
        size_t n;
        n = (size_t)EH_ * G_;
        pack_b_kernel<<<(unsigned)((n + TPB - 1) / TPB), TPB, 0, stream>>>(k0,  k0p,  EH_, G_);
        n = (size_t)H_ * G_;
        pack_b_kernel<<<(unsigned)((n + TPB - 1) / TPB), TPB, 0, stream>>>(rk0, rk0p, H_, G_);
        pack_b_kernel<<<(unsigned)((n + TPB - 1) / TPB), TPB, 0, stream>>>(k1,  k1p,  H_, G_);
        pack_b_kernel<<<(unsigned)((n + TPB - 1) / TPB), TPB, 0, stream>>>(rk1, rk1p, H_, G_);
        n = (size_t)H_ * V_;
        pack_b_kernel<<<(unsigned)((n + TPB - 1) / TPB), TPB, 0, stream>>>(Wd,  Wdp,  H_, V_);
    }

    // waves = (M/16) * (N/128); blocks = waves/8 (256 threads = 8 waves)
    auto gemm_blocks = [](int M, int N) -> unsigned {
        return (unsigned)(((M >> 4) * (N >> 7)) / 8);
    };

    // ---- 3) big precompute: mx0 = xin @ k0 + b0[0]   [BT, 3H] ----
    gemm_bf16_wmma_kernel<<<gemm_blocks(BT_, G_), TPB, 0, stream>>>(
        xin, EH_, k0p, b0, mx0, G_, BT_, G_, EH_);

    // ---- 4) recurrence over T ----
    const unsigned rec_blocks = gemm_blocks(B_, G_);   // (2*24)/8 = 6
    for (int t = 0; t < T_; ++t) {
        // layer 0: mi0 = h0 @ rk0 + b0[1]
        gemm_bf16_wmma_kernel<<<rec_blocks, TPB, 0, stream>>>(
            h0b, H_, rk0p, b0 + G_, mi0, G_, B_, G_, H_);
        // layer 0 gates (mx0 T-slice: row b at (b*T+t) -> ptr + t*G, stride T*G)
        gru_gates_kernel<<<BH_ / TPB, TPB, 0, stream>>>(
            mx0 + (size_t)t * G_, T_ * G_, mi0, h0, h0, h0b, (bf16_t*)nullptr, t);
        // layer 1: mx1 = h0_new @ k1 + b1[0] ; mi1 = h1 @ rk1 + b1[1]
        gemm_bf16_wmma_kernel<<<rec_blocks, TPB, 0, stream>>>(
            h0b, H_, k1p, b1, mx1, G_, B_, G_, H_);
        gemm_bf16_wmma_kernel<<<rec_blocks, TPB, 0, stream>>>(
            h1b, H_, rk1p, b1 + G_, mi1, G_, B_, G_, H_);
        // layer 1 gates -> h1, seq[:, t, :]
        gru_gates_kernel<<<BH_ / TPB, TPB, 0, stream>>>(
            mx1, G_, mi1, h1, h1, h1b, seq, t);
    }

    // ---- 5) projection: logits = seq @ Wd + bd   [BT, V] -> d_out ----
    gemm_bf16_wmma_kernel<<<gemm_blocks(BT_, V_), TPB, 0, stream>>>(
        seq, H_, Wdp, bd, out, V_, BT_, V_, H_);

    // ---- 6) final states ----
    finalize_states_kernel<<<(2 * BH_) / TPB, TPB, 0, stream>>>(
        h0, h1, out + (size_t)BT_ * V_);
}
